// Linear_38482906972841
// MI455X (gfx1250) — matrix operand order repaired and checked by reading, not compiled, not run on hardware
//
#include <hip/hip_runtime.h>

// ---------------------------------------------------------------------------
// y[8192,4096] = x[8192,4096] @ W^T + bias.
// Compute-bound (AI ~820 flop/B) -> bf16 WMMA with bf16x3 split
// (xh*wh + xh*wl + xl*wh) for fp32-class accuracy.
// v2: all operands pre-split to bf16 planes; GEMM inner loop feeds LDS via
// global_load_async_to_lds_b128 (ASYNCcnt) -> no staging VGPRs, no spills.
// ---------------------------------------------------------------------------

typedef __bf16 bf16;
typedef __attribute__((ext_vector_type(16))) __bf16 v16bf;
typedef __attribute__((ext_vector_type(8)))  float  v8f;
typedef __attribute__((ext_vector_type(4)))  __bf16 bf16x4;

#define GM 8192
#define GN 4096
#define GK 4096
#define QIN 4032
#define BM 128
#define BN 128
#define BK 32
#define SK 40                 // LDS row stride (80 B: 16B-aligned, conflict-free)
#define TILE (128 * SK)       // bf16 elems per LDS tile

// ---------------------------------------------------------------------------
// Kernel 1: build permuted bf16 hi/lo weight planes.
// ---------------------------------------------------------------------------
__global__ __launch_bounds__(256) void build_w(
    const float* __restrict__ qw, const float* __restrict__ fw,
    const float* __restrict__ alpha, const float* __restrict__ noise,
    const int* __restrict__ invp,
    unsigned short* __restrict__ whi_u, unsigned short* __restrict__ wlo_u)
{
  int idx = blockIdx.x * 256 + threadIdx.x;   // idx = o*4096 + j
  int o = idx >> 12;
  int j = idx & 4095;
  int i = invp[j];
  float a = alpha[o];
  float val;
  if (i < QIN) {
    float w  = qw[o * QIN + i];
    float wr = noise[o * QIN + i] * 0.5f * (a * (1.0f / 7.0f));
    float c  = (w <= -a) ? -a : (w + wr);
    val = (w >= a) ? a : c;
  } else {
    val = fw[o * 64 + (i - QIN)];
  }
  bf16 h = (bf16)val;
  bf16 l = (bf16)(val - (float)h);
  ((bf16*)whi_u)[idx] = h;
  ((bf16*)wlo_u)[idx] = l;
}

// ---------------------------------------------------------------------------
// Kernel 2: split x (fp32) into bf16 hi/lo planes.
// ---------------------------------------------------------------------------
__global__ __launch_bounds__(256) void split_x(
    const float* __restrict__ x,
    unsigned short* __restrict__ xhi_u, unsigned short* __restrict__ xlo_u)
{
  size_t idx = (size_t)blockIdx.x * 256 + threadIdx.x;   // float4 index
  float4 f = reinterpret_cast<const float4*>(x)[idx];
  bf16x4 hv, lv;
  hv.x = (bf16)f.x; hv.y = (bf16)f.y; hv.z = (bf16)f.z; hv.w = (bf16)f.w;
  lv.x = (bf16)(f.x - (float)hv.x);
  lv.y = (bf16)(f.y - (float)hv.y);
  lv.z = (bf16)(f.z - (float)hv.z);
  lv.w = (bf16)(f.w - (float)hv.w);
  reinterpret_cast<bf16x4*>(xhi_u)[idx] = hv;
  reinterpret_cast<bf16x4*>(xlo_u)[idx] = lv;
}

// ---------------------------------------------------------------------------
// Async-to-LDS helpers (CDNA5, tracked by ASYNCcnt).
// ---------------------------------------------------------------------------
typedef __attribute__((address_space(3))) bf16 lds_bf16;

static __device__ inline unsigned lds_byte_addr(bf16* p) {
  return (unsigned)(unsigned long long)(lds_bf16*)p;   // generic -> LDS offset
}

// Copy one 128x32 bf16 tile (one 16B chunk per lane per instr; 2 instrs/thread).
static __device__ inline void async_tile(const unsigned short* __restrict__ plane,
                                         int rowbase, int kt, bf16* ldsTile, int tid) {
#pragma unroll
  for (int i = 0; i < 2; i++) {
    int c  = tid + i * 256;          // chunk id 0..511
    int r  = c >> 2;                 // tile row 0..127
    int c8 = c & 3;                  // 8-elem column group
    unsigned voff  = ((unsigned)(rowbase + r) * (unsigned)GK +
                      (unsigned)kt + (unsigned)(c8 * 8)) * 2u;  // global byte off
    unsigned laddr = lds_byte_addr(&ldsTile[r * SK + c8 * 8]);
    asm volatile("global_load_async_to_lds_b128 %0, %1, %2"
                 :: "v"(laddr), "v"(voff), "s"(plane) : "memory");
  }
}

static __device__ inline v16bf load_frag2(const bf16* p0, const bf16* p1) {
  v16bf r;
  reinterpret_cast<uint4*>(&r)[0] = *reinterpret_cast<const uint4*>(p0);
  reinterpret_cast<uint4*>(&r)[1] = *reinterpret_cast<const uint4*>(p1);
  return r;
}

// ---------------------------------------------------------------------------
// Kernel 3: GEMM, bf16x3 via v_wmma_f32_16x16x32_bf16, async double-buffer.
// ---------------------------------------------------------------------------
__global__ __launch_bounds__(256) void gemm_bf16x3(
    const unsigned short* __restrict__ xhi_u, const unsigned short* __restrict__ xlo_u,
    const unsigned short* __restrict__ whi_u, const unsigned short* __restrict__ wlo_u,
    const float* __restrict__ bias, float* __restrict__ y)
{
  // 2 stages x {Ah, Al, Bh, Bl} x 128x40 bf16 = 80 KB
  __shared__ alignas(16) bf16 LB[2][4][TILE];

  const int tid  = threadIdx.x;
  const int mblk = blockIdx.y * BM;
  const int nblk = blockIdx.x * BN;

  // wave32 tiling: 8 waves as 2(M) x 4(N); wave tile = 64x32 = 4x2 WMMA tiles
  const int w    = tid >> 5;
  const int lane = tid & 31;
  const int wmi  = w & 1;
  const int wni  = w >> 1;
  const int lm   = lane & 15;
  const int lh   = lane >> 4;

  v8f acc[4][2];
  const v8f vzero = {0.f, 0.f, 0.f, 0.f, 0.f, 0.f, 0.f, 0.f};
#pragma unroll
  for (int i = 0; i < 4; i++)
#pragma unroll
    for (int j = 0; j < 2; j++) acc[i][j] = vzero;

  // prologue: issue stage 0
  async_tile(xhi_u, mblk, 0, &LB[0][0][0], tid);
  async_tile(xlo_u, mblk, 0, &LB[0][1][0], tid);
  async_tile(whi_u, nblk, 0, &LB[0][2][0], tid);
  async_tile(wlo_u, nblk, 0, &LB[0][3][0], tid);

  int buf = 0;
  for (int kt = 0; kt < GK; kt += BK) {
    asm volatile("s_wait_asynccnt 0x0" ::: "memory");  // my stage writes landed
    __syncthreads();                                    // everyone's landed

    if (kt + BK < GK) {                                 // issue next stage
      int nb = buf ^ 1;
      async_tile(xhi_u, mblk, kt + BK, &LB[nb][0][0], tid);
      async_tile(xlo_u, mblk, kt + BK, &LB[nb][1][0], tid);
      async_tile(whi_u, nblk, kt + BK, &LB[nb][2][0], tid);
      async_tile(wlo_u, nblk, kt + BK, &LB[nb][3][0], tid);
    }

    bf16* Ah = &LB[buf][0][0];
    bf16* Al = &LB[buf][1][0];
    bf16* Bh = &LB[buf][2][0];
    bf16* Bl = &LB[buf][3][0];

    // B fragments: 32x16 bf16; V0-3 hold K kb..kb+7, V4-7 hold K 16+kb..16+kb+7
    v16bf bh[2], bl[2];
#pragma unroll
    for (int nt = 0; nt < 2; nt++) {
      int r = wni * 32 + nt * 16 + lm;
      const bf16* p = &Bh[r * SK + lh * 8];
      const bf16* q = &Bl[r * SK + lh * 8];
      bh[nt] = load_frag2(p, p + 16);
      bl[nt] = load_frag2(q, q + 16);
    }

#pragma unroll
    for (int mt = 0; mt < 4; mt++) {
      // A fragment 16x32: V0-3 hold K kb..kb+7, V4-7 hold K 16+kb..16+kb+7
      int r = wmi * 64 + mt * 16 + lm;
      const bf16* p = &Ah[r * SK + lh * 8];
      const bf16* q = &Al[r * SK + lh * 8];
      v16bf ah = load_frag2(p, p + 16);
      v16bf al = load_frag2(q, q + 16);
#pragma unroll
      for (int nt = 0; nt < 2; nt++) {
        acc[mt][nt] = __builtin_amdgcn_wmma_f32_16x16x32_bf16(
            false, ah, false, bh[nt], (short)0, acc[mt][nt], false, false);
        acc[mt][nt] = __builtin_amdgcn_wmma_f32_16x16x32_bf16(
            false, ah, false, bl[nt], (short)0, acc[mt][nt], false, false);
        acc[mt][nt] = __builtin_amdgcn_wmma_f32_16x16x32_bf16(
            false, al, false, bh[nt], (short)0, acc[mt][nt], false, false);
      }
    }
    buf ^= 1;
  }

  // Epilogue: C/D layout -> lane<16: N=lane, M=v; lane>=16: N=lane-16, M=v+8
#pragma unroll
  for (int mt = 0; mt < 4; mt++) {
#pragma unroll
    for (int nt = 0; nt < 2; nt++) {
      int n     = nblk + wni * 32 + nt * 16 + lm;
      int mbase = mblk + wmi * 64 + mt * 16 + lh * 8;
      float bv  = bias[n];
#pragma unroll
      for (int v = 0; v < 8; v++) {
        y[(size_t)(mbase + v) * GN + n] = acc[mt][nt][v] + bv;
      }
    }
  }
}

// ---------------------------------------------------------------------------
extern "C" void kernel_launch(void* const* d_in, const int* in_sizes, int n_in,
                              void* d_out, int out_size, void* d_ws, size_t ws_size,
                              hipStream_t stream) {
  const float* x     = (const float*)d_in[0];
  const float* qw    = (const float*)d_in[1];
  const float* fw    = (const float*)d_in[2];
  const float* alpha = (const float*)d_in[3];
  const float* bias  = (const float*)d_in[4];
  const float* noise = (const float*)d_in[5];
  const int*   invp  = (const int*)d_in[6];
  float*       y     = (float*)d_out;

  // ws layout (bf16 planes): Whi 32MB | Wlo 32MB | Xhi 64MB | Xlo 64MB = 192MB
  unsigned short* whi = (unsigned short*)d_ws;
  unsigned short* wlo = whi + (size_t)GN * GK;
  unsigned short* xhi = wlo + (size_t)GN * GK;
  unsigned short* xlo = xhi + (size_t)GM * GK;

  build_w<<<(GN * GK) / 256, 256, 0, stream>>>(qw, fw, alpha, noise, invp, whi, wlo);
  split_x<<<(GM * GK) / 4 / 256, 256, 0, stream>>>(x, xhi, xlo);

  dim3 grid(GN / BN, GM / BM);  // 32 x 64 blocks
  gemm_bf16x3<<<grid, 256, 0, stream>>>(xhi, xlo, whi, wlo, bias, y);
}
